// GRU_19430432047719
// MI455X (gfx1250) — matrix operand order repaired and checked by reading, not compiled, not run on hardware
//
#include <hip/hip_runtime.h>

// ---------------------------------------------------------------------------
// 2-layer GRU + FC for MI455X (gfx1250).
//  * bf16 WMMA (v_wmma_f32_16x16x32_bf16), f32 accumulate.
//  * Input projections: WMMA GEMM, B tiles staged in LDS via CDNA5 async
//    copies (global_load_async_to_lds_b128 + s_wait_asynccnt), double-buffered
//    and fully unrolled (K is a template constant) so the ping-pong is pure
//    register renaming and LDS offsets are immediates.
//  * Recurrence: one kernel per timestep (graph-captured), fully unrolled
//    register-pipelined K-loop so WMMAs overlap w_hh loads; w_hh (6 MB bf16)
//    stays resident in the 192 MB L2 across all 256 steps.
// ---------------------------------------------------------------------------

#define B_  64
#define T_  256
#define I_  512
#define H_  1024
#define C_  1000
#define G3  (3 * H_)

typedef __bf16 v16bf __attribute__((ext_vector_type(16)));
typedef __bf16 v8bf  __attribute__((ext_vector_type(8)));
typedef float  v8f   __attribute__((ext_vector_type(8)));

__device__ __forceinline__ unsigned short f2bf(float f) {
  unsigned u = __builtin_bit_cast(unsigned, f);
  u += 0x7FFFu + ((u >> 16) & 1u);          // round-to-nearest-even
  return (unsigned short)(u >> 16);
}

__device__ __forceinline__ v16bf combine(v8bf lo, v8bf hi) {
  return __builtin_shufflevector(lo, hi, 0, 1, 2, 3, 4, 5, 6, 7,
                                 8, 9, 10, 11, 12, 13, 14, 15);
}

__device__ __forceinline__ v16bf ldfrag(const __bf16* p0, const __bf16* p1) {
  return combine(*(const v8bf*)p0, *(const v8bf*)p1);
}

__device__ __forceinline__ v8f wmma_bf16(v16bf a, v16bf b, v8f c) {
  return __builtin_amdgcn_wmma_f32_16x16x32_bf16(
      false, a, false, b, (short)0, c, false, false);
}

__device__ __forceinline__ float sigmoidf_(float x) {
  return 1.0f / (1.0f + __expf(-x));
}

// Low 32 bits of a flat pointer into the LDS aperture == LDS byte offset.
__device__ __forceinline__ unsigned lds_off_of(const void* p) {
  return (unsigned)(unsigned long long)p;
}

__device__ __forceinline__ void async_ld_b128(unsigned lds_off, const void* g) {
  asm volatile("global_load_async_to_lds_b128 %0, %1, off"
               :: "v"(lds_off), "v"(g) : "memory");
}

__device__ __forceinline__ void wait_async0() {
  asm volatile("s_wait_asynccnt 0x0" ::: "memory");
}

// --------------------------- f32 -> bf16 conversion -------------------------
__global__ void cvt_f32_bf16(const float* __restrict__ in,
                             __bf16* __restrict__ out, size_t n) {
  size_t i = (size_t)blockIdx.x * blockDim.x + threadIdx.x;
  size_t stride = (size_t)gridDim.x * blockDim.x;
  unsigned short* o = (unsigned short*)out;
  for (; i < n; i += stride) o[i] = f2bf(in[i]);
}

// ------------------------------ zero h state -------------------------------
__global__ void zero_h(float* __restrict__ hf, __bf16* __restrict__ hb) {
  int i = blockIdx.x * 256 + threadIdx.x;   // grid 256 x 256 == B_*H_ threads
  hf[i] = 0.0f;
  ((unsigned short*)hb)[i] = 0;
}

// ------------------- OUT(M,N) = A(M,K) * W(N,K)^T + bias -------------------
// Block = 8 waves, tile 128M x 64N. The shared 64x32 B tile is staged in LDS
// with async global->LDS copies (double-buffered); A fragments are register
// double-buffered. Rows padded to 40 elements (80 B) for bank spread.
// K is compile-time so the whole pipeline is fully unrolled.
template <int K>
__global__ __launch_bounds__(256) void gemm_bf16_wmma(
    const __bf16* __restrict__ A, const __bf16* __restrict__ W,
    const float* __restrict__ bias, float* __restrict__ OUT,
    int M, int N) {
  __shared__ __bf16 smB[2][64][40];

  const int lane  = threadIdx.x & 31;
  const int wave  = threadIdx.x >> 5;
  const int m0    = blockIdx.y * 128 + wave * 16;
  const int n_base = blockIdx.x * 64;
  const int lhalf  = lane >> 4;
  const int lmod   = lane & 15;

  // Async B-tile fill: one 16-byte chunk per thread (64 rows x 4 segments).
  const int frow = threadIdx.x >> 2;        // 0..63 : n within tile
  const int fseg = threadIdx.x & 3;         // 0..3  : 16B segment along K
  int fnr = n_base + frow; if (fnr > N - 1) fnr = N - 1;
  const __bf16* fgbase = W + (size_t)fnr * K + fseg * 8;

  v8f acc[4] = {v8f{}, v8f{}, v8f{}, v8f{}};

  int marow = m0 + lmod; if (marow > M - 1) marow = M - 1;
  const __bf16* Abase = A + (size_t)marow * K + (lhalf << 3);
  const int kofsB = lhalf << 3;

  constexpr int nk = K / 32;

  // Prologue: issue buf0 fill, load first A fragment.
  async_ld_b128(lds_off_of(&smB[0][frow][fseg * 8]), fgbase);
  v16bf a_cur = ldfrag(Abase, Abase + 16);
  wait_async0();
  __syncthreads();

#pragma unroll
  for (int it = 0; it < nk; ++it) {
    const int cur = it & 1;
    const int k0 = it * 32;
    const bool more = (it + 1 < nk);

    v16bf a_nxt = a_cur;
    if (more) {
      async_ld_b128(lds_off_of(&smB[cur ^ 1][frow][fseg * 8]),
                    fgbase + (k0 + 32));
      a_nxt = ldfrag(Abase + k0 + 32, Abase + k0 + 48);
    }

    // Hoist all four B-fragment LDS reads before the WMMAs so the compiler
    // can use partial dscnt waits instead of draining before each one.
    const __bf16* bp0 = &smB[cur][ 0 + lmod][kofsB];
    const __bf16* bp1 = &smB[cur][16 + lmod][kofsB];
    const __bf16* bp2 = &smB[cur][32 + lmod][kofsB];
    const __bf16* bp3 = &smB[cur][48 + lmod][kofsB];
    v16bf b0 = ldfrag(bp0, bp0 + 16);
    v16bf b1 = ldfrag(bp1, bp1 + 16);
    v16bf b2 = ldfrag(bp2, bp2 + 16);
    v16bf b3 = ldfrag(bp3, bp3 + 16);
    acc[0] = wmma_bf16(a_cur, b0, acc[0]);
    acc[1] = wmma_bf16(a_cur, b1, acc[1]);
    acc[2] = wmma_bf16(a_cur, b2, acc[2]);
    acc[3] = wmma_bf16(a_cur, b3, acc[3]);

    if (more) {
      wait_async0();
      __syncthreads();
      a_cur = a_nxt;
    }
  }

#pragma unroll
  for (int i = 0; i < 4; ++i) {
    const int n = n_base + i * 16 + lmod;
    if (n < N) {
      const float bv = bias[n];
#pragma unroll
      for (int p = 0; p < 8; ++p) {
        const int m = m0 + p + (lhalf << 3);
        if (m < M) OUT[(size_t)m * N + n] = acc[i][p] + bv;
      }
    }
  }
}

// ------------------------------- GRU timestep ------------------------------
// One launch per t. 32 blocks x 8 waves = 256 waves; wave -> one 16x16 h tile
// with all three gate tiles (r,z,n). Fully unrolled register-pipelined K-loop:
// next-step fragments are issued before the current WMMAs so loads overlap
// matrix ops; full unroll turns buffer swaps into register renaming.
__global__ __launch_bounds__(256) void gru_step(
    const __bf16* __restrict__ h_in_bf, const float* __restrict__ h_in_f,
    const __bf16* __restrict__ w_hh, const float* __restrict__ b_hh,
    const float* __restrict__ xp, int t,
    float* __restrict__ h_out_f, __bf16* __restrict__ h_out_bf,
    __bf16* __restrict__ y /* may be null (layer 1) */) {
  const int lane = threadIdx.x & 31;
  const int wave = threadIdx.x >> 5;
  const int id   = blockIdx.x * 8 + wave;    // 0..255
  const int m0   = (id >> 6) << 4;           // batch-tile base   (4 tiles)
  const int j0   = (id & 63) << 4;           // hidden-tile base  (64 tiles)
  const int lhalf = lane >> 4;
  const int lmod  = lane & 15;

  v8f accr = {}, accz = {}, accn = {};

  const __bf16* Abase = h_in_bf + (size_t)(m0 + lmod) * H_ + (lhalf << 3);
  const int kofsB = lhalf << 3;
  const __bf16* Br = w_hh + (size_t)(j0 + lmod) * H_ + kofsB;
  const __bf16* Bz = Br + (size_t)H_ * H_;
  const __bf16* Bn = Bz + (size_t)H_ * H_;

  v16bf a  = ldfrag(Abase, Abase + 16);
  v16bf br = ldfrag(Br, Br + 16);
  v16bf bz = ldfrag(Bz, Bz + 16);
  v16bf bn = ldfrag(Bn, Bn + 16);

#pragma unroll
  for (int k0 = 0; k0 < H_ - 32; k0 += 32) {
    const int kn = k0 + 32;
    v16bf a2  = ldfrag(Abase + kn, Abase + kn + 16);
    v16bf br2 = ldfrag(Br + kn, Br + kn + 16);
    v16bf bz2 = ldfrag(Bz + kn, Bz + kn + 16);
    v16bf bn2 = ldfrag(Bn + kn, Bn + kn + 16);
    accr = wmma_bf16(a, br, accr);
    accz = wmma_bf16(a, bz, accz);
    accn = wmma_bf16(a, bn, accn);
    a = a2; br = br2; bz = bz2; bn = bn2;
  }
  accr = wmma_bf16(a, br, accr);
  accz = wmma_bf16(a, bz, accz);
  accn = wmma_bf16(a, bn, accn);

  const int n = j0 + lmod;
  const float bhr = b_hh[n];
  const float bhz = b_hh[H_ + n];
  const float bhn = b_hh[2 * H_ + n];

#pragma unroll
  for (int p = 0; p < 8; ++p) {
    const int m = m0 + p + (lhalf << 3);     // batch index
    const float* xrow = xp + ((size_t)m * T_ + t) * (size_t)G3;
    const float xr = xrow[n];
    const float xz = xrow[H_ + n];
    const float xn = xrow[2 * H_ + n];
    const float r  = sigmoidf_(xr + accr[p] + bhr);
    const float z  = sigmoidf_(xz + accz[p] + bhz);
    const float nv = tanhf(xn + r * (accn[p] + bhn));
    const float hold = h_in_f[(size_t)m * H_ + n];
    const float hnew = (1.0f - z) * nv + z * hold;
    h_out_f[(size_t)m * H_ + n] = hnew;
    ((unsigned short*)h_out_bf)[(size_t)m * H_ + n] = f2bf(hnew);
    if (y) ((unsigned short*)y)[((size_t)m * T_ + t) * H_ + n] = f2bf(hnew);
  }
}

// ------------------------------- launch ------------------------------------
extern "C" void kernel_launch(void* const* d_in, const int* in_sizes, int n_in,
                              void* d_out, int out_size, void* d_ws,
                              size_t ws_size, hipStream_t stream) {
  (void)in_sizes; (void)n_in; (void)out_size; (void)ws_size;

  const float* x    = (const float*)d_in[0];
  const float* wih0 = (const float*)d_in[1];
  const float* whh0 = (const float*)d_in[2];
  const float* bih0 = (const float*)d_in[3];
  const float* bhh0 = (const float*)d_in[4];
  const float* wih1 = (const float*)d_in[5];
  const float* whh1 = (const float*)d_in[6];
  const float* bih1 = (const float*)d_in[7];
  const float* bhh1 = (const float*)d_in[8];
  const float* fcw  = (const float*)d_in[9];
  const float* fcb  = (const float*)d_in[10];
  float* out = (float*)d_out;

  char* w = (char*)d_ws;
  auto alloc = [&](size_t bytes) {
    char* p = w;
    w += (bytes + 255) & ~(size_t)255;
    return p;
  };
  __bf16* x_bf    = (__bf16*)alloc((size_t)B_ * T_ * I_ * 2);
  __bf16* wih0_bf = (__bf16*)alloc((size_t)G3 * I_ * 2);
  __bf16* whh0_bf = (__bf16*)alloc((size_t)G3 * H_ * 2);
  __bf16* wih1_bf = (__bf16*)alloc((size_t)G3 * H_ * 2);
  __bf16* whh1_bf = (__bf16*)alloc((size_t)G3 * H_ * 2);
  __bf16* fcw_bf  = (__bf16*)alloc((size_t)C_ * H_ * 2);
  float*  xp      = (float*)alloc((size_t)B_ * T_ * G3 * 4);
  __bf16* y0_bf   = (__bf16*)alloc((size_t)B_ * T_ * H_ * 2);
  float*  hf0     = (float*)alloc((size_t)B_ * H_ * 4);
  float*  hf1     = (float*)alloc((size_t)B_ * H_ * 4);
  __bf16* hb0     = (__bf16*)alloc((size_t)B_ * H_ * 2);
  __bf16* hb1     = (__bf16*)alloc((size_t)B_ * H_ * 2);
  float*  hf[2]   = {hf0, hf1};
  __bf16* hb[2]   = {hb0, hb1};

  auto cvt = [&](const float* s, __bf16* d, size_t n) {
    cvt_f32_bf16<<<2048, 256, 0, stream>>>(s, d, n);
  };
  cvt(x,    x_bf,    (size_t)B_ * T_ * I_);
  cvt(wih0, wih0_bf, (size_t)G3 * I_);
  cvt(whh0, whh0_bf, (size_t)G3 * H_);
  cvt(wih1, wih1_bf, (size_t)G3 * H_);
  cvt(whh1, whh1_bf, (size_t)G3 * H_);
  cvt(fcw,  fcw_bf,  (size_t)C_ * H_);

  // Layer 0: xp = x @ w_ih0^T + b_ih0   (M=16384, N=3072, K=512)
  gemm_bf16_wmma<I_><<<dim3(G3 / 64, (B_ * T_) / 128), 256, 0, stream>>>(
      x_bf, wih0_bf, bih0, xp, B_ * T_, G3);
  zero_h<<<256, 256, 0, stream>>>(hf[0], hb[0]);
  for (int t = 0; t < T_; ++t) {
    const int i = t & 1, o = i ^ 1;
    gru_step<<<32, 256, 0, stream>>>(hb[i], hf[i], whh0_bf, bhh0, xp, t,
                                     hf[o], hb[o], y0_bf);
  }

  // Layer 1: xp = y0 @ w_ih1^T + b_ih1  (M=16384, N=3072, K=1024)
  gemm_bf16_wmma<H_><<<dim3(G3 / 64, (B_ * T_) / 128), 256, 0, stream>>>(
      y0_bf, wih1_bf, bih1, xp, B_ * T_, G3);
  zero_h<<<256, 256, 0, stream>>>(hf[0], hb[0]);
  for (int t = 0; t < T_; ++t) {
    const int i = t & 1, o = i ^ 1;
    gru_step<<<32, 256, 0, stream>>>(hb[i], hf[i], whh1_bf, bhh1, xp, t,
                                     hf[o], hb[o], (__bf16*)nullptr);
  }

  // FC: out = h_last @ fc_w^T + fc_b    (M=64, N=1000, K=1024)
  // After t = 255 (odd), final h lives in slot 0.
  gemm_bf16_wmma<H_><<<dim3((C_ + 63) / 64, 1), 256, 0, stream>>>(
      hb[0], fcw_bf, fcb, out, B_, C_);
}
